// MPNN_30039001269034
// MI455X (gfx1250) — matrix operand order repaired and checked by reading, not compiled, not run on hardware
//
#include <hip/hip_runtime.h>
#include <hip/hip_bf16.h>

// ---- problem constants (match reference) ----
static constexpr int B_    = 4096;
static constexpr int N_    = 64;
static constexpr int FEAT_ = 32;
static constexpr int H_    = 128;
static constexpr int E_    = 4;
static constexpr int L_    = 3;

typedef __attribute__((ext_vector_type(16))) _Float16 v16h;
typedef __attribute__((ext_vector_type(8)))  _Float16 v8h;
typedef __attribute__((ext_vector_type(8)))  float    v8f;

// ---------------- Tensor Data Mover (TDM) 1-D LDS fill ----------------
#if __has_builtin(__builtin_amdgcn_tensor_load_to_lds)
#define HAVE_TDM 1
typedef __attribute__((ext_vector_type(4))) unsigned int u32x4_t;
typedef __attribute__((ext_vector_type(8))) int          i32x8_t;
typedef __attribute__((ext_vector_type(4))) int          i32x4_t;

// Copy n8 8-byte elements from global to LDS via the TDM (tracked by TENSORcnt).
// D# per CDNA5 ISA ch.8: group0 = {count=1 | lds_addr | global_addr | type=2},
// group1 = {data_size=8B, tensor_dim0 = tile_dim0 = n8}, groups 2/3 zero (<=2D).
__device__ __forceinline__ void tdm_copy_to_lds(void* ldsDst, const void* gSrc,
                                                unsigned n8) {
  unsigned lds           = (unsigned)(uintptr_t)ldsDst;   // LDS byte address
  unsigned long long ga  = (unsigned long long)(uintptr_t)gSrc;
  u32x4_t g0;
  g0[0] = 1u;                                            // one valid descriptor
  g0[1] = lds;
  g0[2] = (unsigned)ga;                                  // global_addr[31:0]
  g0[3] = (unsigned)((ga >> 32) & 0x1ffffffu) | (2u << 30); // [56:32] | type=2
  i32x8_t g1;
  g1[0] = (3 << 16);                                     // data_size = 8 bytes
  g1[1] = (int)((n8 & 0xffffu) << 16);                   // tensor_dim0[15:0]
  g1[2] = (int)(n8 >> 16);                               // tensor_dim0[31:16]
  g1[3] = (int)((n8 & 0xffffu) << 16);                   // tile_dim0 = n8
  g1[4] = 0; g1[5] = 0; g1[6] = 0; g1[7] = 0;
  i32x4_t gz;
  gz[0] = 0; gz[1] = 0; gz[2] = 0; gz[3] = 0;
#if defined(__clang_major__) && __clang_major__ >= 23
  i32x8_t gz8;
  gz8[0] = 0; gz8[1] = 0; gz8[2] = 0; gz8[3] = 0;
  gz8[4] = 0; gz8[5] = 0; gz8[6] = 0; gz8[7] = 0;
  __builtin_amdgcn_tensor_load_to_lds(g0, g1, gz, gz, gz8, 0);
#else
  __builtin_amdgcn_tensor_load_to_lds(g0, g1, gz, gz, 0);
#endif
}
#endif

// ---------------- WMMA helpers (wave32, 16x16x32 f16 -> f32) ----------------

__device__ __forceinline__ v8f wmma_f16(v16h a, v16h b, v8f c) {
  return __builtin_amdgcn_wmma_f32_16x16x32_f16(false, a, false, b, (short)0, c,
                                                false, false);
}

// A fragment: 16(M) x 32(K) f16 from f16 row-major (ld halves/row), 16B loads.
__device__ __forceinline__ v16h load_a_h(const _Float16* src, int ld) {
  int lane = threadIdx.x & 31;
  int row  = lane & 15;
  int hi   = lane >> 4;
  const _Float16* p = src + (size_t)row * ld + hi * 16;
  v8h lo = *(const v8h*)p;          // K k..k+7
  v8h hh = *(const v8h*)(p + 8);    // K k+8..k+15
  v16h a;
#pragma unroll
  for (int t = 0; t < 8; ++t) { a[t] = lo[t]; a[8 + t] = hh[t]; }
  return a;
}

// A fragment from f32 row-major global (x, state).
__device__ __forceinline__ v16h load_a_f32(const float* __restrict__ src, int ld) {
  int lane = threadIdx.x & 31;
  int row  = lane & 15;
  int hi   = lane >> 4;
  const float* p = src + (size_t)row * ld + hi * 16;
  v16h a;
#pragma unroll
  for (int t = 0; t < 8; ++t) a[t]     = (_Float16)p[t];
#pragma unroll
  for (int t = 0; t < 8; ++t) a[8 + t] = (_Float16)p[8 + t];
  return a;
}

// Packed-B layout: a (K x N) matrix stored as (K/32)x(N/16) panels of
// 32 lanes x 16 halves (1KB). Lane = kHi*16 + col holds K = kp*32+kHi*16+t.
__device__ __forceinline__ v16h load_b_packed(const _Float16* base) {
  int lane = threadIdx.x & 31;
  const _Float16* p = base + lane * 16;
  v8h lo = *(const v8h*)p;
  v8h hh = *(const v8h*)(p + 8);
  v16h b;
#pragma unroll
  for (int t = 0; t < 8; ++t) { b[t] = lo[t]; b[8 + t] = hh[t]; }
  return b;
}

// C/D 16x16 f32 tile: VGPR r -> rows r (lanes 0..15) / r+8 (lanes 16..31).
__device__ __forceinline__ void store_c_f32(float* __restrict__ dst, int ld, v8f c) {
  int lane = threadIdx.x & 31;
  int col  = lane & 15;
  int hi   = lane >> 4;
#pragma unroll
  for (int r = 0; r < 8; ++r) dst[(size_t)(r + hi * 8) * ld + col] = c[r];
}

// Store a C tile (rows mt*16.., cols nt*16..) of the 64x128 support matrix S
// directly into packed-B layout (K=64 -> 2 row-panels, N=128 -> NP=8).
__device__ __forceinline__ void store_c_packedB(_Float16* base, int mt, int nt,
                                                v8f c) {
  int lane = threadIdx.x & 31;
  int col  = lane & 15;
  int hi   = lane >> 4;
#pragma unroll
  for (int r = 0; r < 8; ++r) {
    int j     = mt * 16 + r + hi * 8;       // S row = K index (0..63)
    int kp    = j >> 5;
    int jj    = j & 31;
    int lane2 = (jj >> 4) * 16 + col;
    base[(((size_t)kp * 8 + nt) * 32 + lane2) * 16 + (jj & 15)] = (_Float16)c[r];
  }
}

__device__ __forceinline__ void accum_relu_lds(float* dst, int ld, v8f c) {
  int lane = threadIdx.x & 31;
  int col  = lane & 15;
  int hi   = lane >> 4;
#pragma unroll
  for (int r = 0; r < 8; ++r) {
    float v = c[r];
    v = v > 0.f ? v : 0.f;
    dst[(size_t)(r + hi * 8) * ld + col] += v;
  }
}

__device__ __forceinline__ float sigmoidf_(float x) {
  return 1.0f / (1.0f + __expf(-x));
}

// ---------------- Kernel 0a: streaming f32 -> f16 conversion ----------------
__global__ __launch_bounds__(256) void k_cvt(const float* __restrict__ src,
                                             _Float16* __restrict__ dst, int n) {
  int i = (blockIdx.x * 256 + threadIdx.x) * 8;
  if (i + 8 <= n) {
    float4 a = *(const float4*)(src + i);
    float4 b = *(const float4*)(src + i + 4);
    v8h o;
    o[0] = (_Float16)a.x; o[1] = (_Float16)a.y;
    o[2] = (_Float16)a.z; o[3] = (_Float16)a.w;
    o[4] = (_Float16)b.x; o[5] = (_Float16)b.y;
    o[6] = (_Float16)b.z; o[7] = (_Float16)b.w;
    *(v8h*)(dst + i) = o;
  }
}

// ---------------- Kernel 0b: pack f32 weights into WMMA B-fragment layout ---
// Compile-time shapes -> divisions become shifts/constant magic.
// !TRANS: src is (K x Nn) row-major, element (k,n)=src[k*Nn+n]
//  TRANS: src is (Nn x K) row-major, element (k,n)=src[n*K+k]
template <int K, int Nn, int TRANS>
__global__ __launch_bounds__(256) void k_pack_b(const float* __restrict__ src,
                                                _Float16* __restrict__ dst,
                                                int nmat) {
  constexpr int NP     = Nn >> 4;
  constexpr int perMat = (K >> 5) * NP * 32;
  int idx = blockIdx.x * 256 + threadIdx.x;
  int mat = idx / perMat;
  if (mat >= nmat) return;
  int rem   = idx % perMat;
  int lane  = rem & 31;
  int panel = rem >> 5;
  int kp = panel / NP, np = panel % NP;
  int col = lane & 15, hi = lane >> 4;
  int k0  = kp * 32 + hi * 16;
  int n   = np * 16 + col;
  const float* s = src + (size_t)mat * K * Nn;
  _Float16*    d = dst + (size_t)mat * K * Nn + ((size_t)panel * 32 + lane) * 16;
  v8h o0, o1;
#pragma unroll
  for (int t = 0; t < 8; ++t) {
    float v0 = TRANS ? s[(size_t)n * K + k0 + t]
                     : s[(size_t)(k0 + t) * Nn + n];
    float v1 = TRANS ? s[(size_t)n * K + k0 + 8 + t]
                     : s[(size_t)(k0 + 8 + t) * Nn + n];
    o0[t] = (_Float16)v0;
    o1[t] = (_Float16)v1;
  }
  *(v8h*)d       = o0;
  *(v8h*)(d + 8) = o1;
}

// ---------------- Kernel 1: per-node input embedding ----------------
// state[b,n,:] = x[b,n,:] @ in_w[n] + in_b[n]   (K = FEAT = 32, one WMMA)
__global__ __launch_bounds__(256) void k_embed(const float* __restrict__ x,
                                               const _Float16* __restrict__ in_wP,
                                               const float* __restrict__ in_b,
                                               float* __restrict__ state) {
  int blk  = blockIdx.x;
  int node = blk % N_;
  int bt   = blk / N_;
  int wave = threadIdx.x >> 5;     // 8 waves -> 8 H-tiles of 16
  int b0   = bt * 16;

  v16h a = load_a_f32(x + ((size_t)b0 * N_ + node) * FEAT_, N_ * FEAT_);
  v16h b = load_b_packed(in_wP + (size_t)node * FEAT_ * H_ + (size_t)wave * 512);

  int lane = threadIdx.x & 31;
  float bias = in_b[node * H_ + wave * 16 + (lane & 15)];
  v8f c;
#pragma unroll
  for (int r = 0; r < 8; ++r) c[r] = bias;

  c = wmma_f16(a, b, c);
  store_c_f32(state + ((size_t)b0 * N_ + node) * H_ + wave * 16, N_ * H_, c);
}

// ---------------- Kernel 2: fused R-GCN layer + GRU cell ----------------
// One block per graph b (64 nodes x 128 hidden). 8 waves. LDS = 104 KB.
__global__ __launch_bounds__(256) void k_layer(
    const float* __restrict__ stateIn,     // (B*N, H) f32 (== hidden == cur)
    const _Float16* __restrict__ adjH,     // (B, E, N, N) f16
    const _Float16* __restrict__ gc_wP,    // (E, H*H) f16 packed-B (layer slice)
    const float* __restrict__ gc_b,        // (E, H) f32 (layer slice)
    const _Float16* __restrict__ w_ihP,    // (3H x H) f16 packed-B (transposed)
    const _Float16* __restrict__ w_hhP,    // (3H x H) f16 packed-B (transposed)
    const float* __restrict__ b_ih,        // (3H,)
    const float* __restrict__ b_hh,        // (3H,)
    float* __restrict__ stateOut) {        // (B*N, H) f32
  __shared__ __align__(16) _Float16 curH[N_ * H_];   // 16 KB hidden/cur f16 (A)
  __shared__ __align__(16) _Float16 sS[N_ * H_];     // 16 KB S packed-B, then agg
  __shared__ __align__(16) _Float16 sW[H_ * H_];     // 32 KB staged gc_w[e] packed
  __shared__ __align__(16) _Float16 sAdj[N_ * N_];   //  8 KB staged adj[b,e] (A)
  __shared__ __align__(16) float    sAgg[N_ * H_];   // 32 KB f32 edge accumulator

  int b    = blockIdx.x;
  int tid  = threadIdx.x;
  int wave = tid >> 5;
  int lane = tid & 31;
  int colL = lane & 15;
  int hiL  = lane >> 4;

  const float* curG = stateIn + (size_t)b * N_ * H_;

  for (int i = tid; i < N_ * H_; i += 256) {
    curH[i] = (_Float16)curG[i];
    sAgg[i] = 0.f;
  }
  __syncthreads();

  // ---- relational GCN: agg = mean_e relu(adj[b,e] @ (cur @ W_e) + bias_e) ----
  for (int e = 0; e < E_; ++e) {
#if defined(HAVE_TDM)
    // DMA the packed W_e slab (32 KB) and adj[b,e] slab (8 KB) into LDS.
    if (wave == 0) {
      tdm_copy_to_lds(sW, gc_wP + (size_t)e * H_ * H_, (H_ * H_ * 2) / 8);
      tdm_copy_to_lds(sAdj, adjH + ((size_t)b * E_ + e) * N_ * N_,
                      (N_ * N_ * 2) / 8);
#if __has_builtin(__builtin_amdgcn_s_wait_tensorcnt)
      __builtin_amdgcn_s_wait_tensorcnt(0);
#endif
    }
#else
    {
      const uint4* wg = (const uint4*)(gc_wP + (size_t)e * H_ * H_);
      uint4* wl = (uint4*)sW;
#pragma unroll
      for (int i = 0; i < (H_ * H_) / 8 / 256; ++i)
        wl[tid + i * 256] = wg[tid + i * 256];
      const uint4* ag = (const uint4*)(adjH + ((size_t)b * E_ + e) * N_ * N_);
      uint4* al = (uint4*)sAdj;
#pragma unroll
      for (int i = 0; i < (N_ * N_) / 8 / 256; ++i)
        al[tid + i * 256] = ag[tid + i * 256];
    }
#endif
    if (e + 1 < E_)
      __builtin_prefetch(adjH + ((size_t)b * E_ + e + 1) * N_ * N_ + tid * 16, 0, 0);
    __syncthreads();

    // S = curH @ W_e : 64x128 = 32 tiles, 4 per wave, K = 128
    for (int t = wave; t < 32; t += 8) {
      int mt = t >> 3, nt = t & 7;
      v8f acc;
#pragma unroll
      for (int r = 0; r < 8; ++r) acc[r] = 0.f;
#pragma unroll
      for (int k = 0; k < H_; k += 32) {
        v16h a  = load_a_h(curH + mt * 16 * H_ + k, H_);
        v16h bb = load_b_packed(sW + (((size_t)(k >> 5) * 8 + nt) * 512));
        acc = wmma_f16(a, bb, acc);
      }
      store_c_packedB(sS, mt, nt, acc);     // S stored in packed-B layout
    }
    __syncthreads();

    // T = adj[b,e] @ S + bias; relu; accumulate into sAgg.  K = 64
    for (int t = wave; t < 32; t += 8) {
      int mt = t >> 3, nt = t & 7;
      float bias = gc_b[e * H_ + nt * 16 + colL];
      v8f acc;
#pragma unroll
      for (int r = 0; r < 8; ++r) acc[r] = bias;
#pragma unroll
      for (int k = 0; k < N_; k += 32) {
        v16h a  = load_a_h(sAdj + (size_t)mt * 16 * N_ + k, N_);
        v16h bb = load_b_packed(sS + (((size_t)(k >> 5) * 8 + nt) * 512));
        acc = wmma_f16(a, bb, acc);
      }
      accum_relu_lds(sAgg + mt * 16 * H_ + nt * 16, H_, acc);
    }
    __syncthreads();
  }

  // mean over edge types; re-use sS (row-major now) as f16 agg A-source
  for (int i = tid; i < N_ * H_; i += 256)
    sS[i] = (_Float16)(sAgg[i] * (1.0f / E_));
  __syncthreads();

  // ---- GRU cell, fused in-register: 6 accumulator tiles per output tile ----
  float* outG = stateOut + (size_t)b * N_ * H_;
  for (int t = wave; t < 32; t += 8) {
    int mt = t >> 3, nt = t & 7;
    int c0 = nt * 16;
    v8f ir, iz, inn, hr, hz, hn;
    float bir = b_ih[c0 + colL],          bhr = b_hh[c0 + colL];
    float biz = b_ih[H_ + c0 + colL],     bhz = b_hh[H_ + c0 + colL];
    float bin = b_ih[2 * H_ + c0 + colL], bhn = b_hh[2 * H_ + c0 + colL];
#pragma unroll
    for (int r = 0; r < 8; ++r) {
      ir[r] = bir; iz[r] = biz; inn[r] = bin;
      hr[r] = bhr; hz[r] = bhz; hn[r]  = bhn;
    }
    // w_ih/w_hh packed panels: kp = k/32 (4), NP = 24, np = gate*8 + nt
#pragma unroll
    for (int k = 0; k < H_; k += 32) {
      size_t kp24 = (size_t)(k >> 5) * 24;
      v16h ag = load_a_h(sS + mt * 16 * H_ + k, H_);    // agg rows
      v16h hh = load_a_h(curH + mt * 16 * H_ + k, H_);  // hidden rows
      v16h br = load_b_packed(w_ihP + (kp24 + 0 * 8 + nt) * 512);
      v16h bz = load_b_packed(w_ihP + (kp24 + 1 * 8 + nt) * 512);
      v16h bn = load_b_packed(w_ihP + (kp24 + 2 * 8 + nt) * 512);
      ir  = wmma_f16(ag, br, ir);
      iz  = wmma_f16(ag, bz, iz);
      inn = wmma_f16(ag, bn, inn);
      v16h cr = load_b_packed(w_hhP + (kp24 + 0 * 8 + nt) * 512);
      v16h cz = load_b_packed(w_hhP + (kp24 + 1 * 8 + nt) * 512);
      v16h cn = load_b_packed(w_hhP + (kp24 + 2 * 8 + nt) * 512);
      hr = wmma_f16(hh, cr, hr);
      hz = wmma_f16(hh, cz, hz);
      hn = wmma_f16(hh, cn, hn);
    }
#pragma unroll
    for (int r = 0; r < 8; ++r) {
      int m  = mt * 16 + r + hiL * 8;
      int cc = c0 + colL;
      float rg = sigmoidf_(ir[r] + hr[r]);
      float zg = sigmoidf_(iz[r] + hz[r]);
      float ng = tanhf(inn[r] + rg * hn[r]);
      float hv = curG[(size_t)m * H_ + cc];
      outG[(size_t)m * H_ + cc] = (1.f - zg) * ng + zg * hv;
    }
  }
}

// ---------------- Kernel 3: per-node output embedding + zero pad ----------------
__global__ __launch_bounds__(64) void k_out(const float* __restrict__ state,
                                            const _Float16* __restrict__ out_wP,
                                            const float* __restrict__ out_b,
                                            float* __restrict__ out) {
  int blk  = blockIdx.x;
  int node = blk % N_;
  int bt   = blk / N_;
  int wave = threadIdx.x >> 5;   // 2 waves -> FEAT tiles 0..1
  int b0   = bt * 16;
  int lane = threadIdx.x & 31;
  int colL = lane & 15;
  int hiL  = lane >> 4;

  const float* Abase = state + ((size_t)b0 * N_ + node) * H_;
  int ldA = N_ * H_;

  v8f acc;
  float bias = out_b[node * FEAT_ + wave * 16 + colL];
#pragma unroll
  for (int r = 0; r < 8; ++r) acc[r] = bias;
  // out_w[node]: (K=128 x N=32) packed -> panels (kp 0..3) x (np 0..1)
#pragma unroll
  for (int k = 0; k < H_; k += 32) {
    v16h a  = load_a_f32(Abase + k, ldA);
    v16h bb = load_b_packed(out_wP + (size_t)node * H_ * FEAT_ +
                            (((size_t)(k >> 5) * 2 + wave) * 512));
    acc = wmma_f16(a, bb, acc);
  }

  float* Obase = out + ((size_t)b0 * N_ + node) * H_;
#pragma unroll
  for (int r = 0; r < 8; ++r)
    Obase[(size_t)(r + hiL * 8) * ldA + wave * 16 + colL] = acc[r];

  // zero-pad columns FEAT..H-1 for the 16 rows this block owns
  for (int i = threadIdx.x; i < 16 * (H_ - FEAT_); i += 64) {
    int rr = i / (H_ - FEAT_);
    int cc = FEAT_ + (i % (H_ - FEAT_));
    Obase[(size_t)rr * ldA + cc] = 0.f;
  }
}

// ---------------- host orchestration ----------------
extern "C" void kernel_launch(void* const* d_in, const int* in_sizes, int n_in,
                              void* d_out, int out_size, void* d_ws, size_t ws_size,
                              hipStream_t stream) {
  const float* x     = (const float*)d_in[0];
  const float* adj   = (const float*)d_in[1];
  const float* in_w  = (const float*)d_in[2];
  const float* in_b  = (const float*)d_in[3];
  const float* out_w = (const float*)d_in[4];
  const float* out_b = (const float*)d_in[5];
  const float* gc_w  = (const float*)d_in[6];
  const float* gc_b  = (const float*)d_in[7];
  const float* w_ih  = (const float*)d_in[8];
  const float* w_hh  = (const float*)d_in[9];
  const float* b_ih  = (const float*)d_in[10];
  const float* b_hh  = (const float*)d_in[11];

  float* out = (float*)d_out;   // (B*N, H) f32, doubles as one state buffer

  // workspace layout (256B aligned slabs)
  char* wp = (char*)d_ws;
  auto take = [&](size_t bytes) {
    char* p = wp;
    wp += (bytes + 255) & ~(size_t)255;
    return p;
  };
  float*    stateA = (float*)take((size_t)B_ * N_ * H_ * 4);         // 128 MiB
  _Float16* adjH   = (_Float16*)take((size_t)B_ * E_ * N_ * N_ * 2); // 128 MiB (L2)
  _Float16* gc_wP  = (_Float16*)take((size_t)L_ * E_ * H_ * H_ * 2);
  _Float16* w_ihP  = (_Float16*)take((size_t)3 * H_ * H_ * 2);
  _Float16* w_hhP  = (_Float16*)take((size_t)3 * H_ * H_ * 2);
  _Float16* in_wP  = (_Float16*)take((size_t)N_ * FEAT_ * H_ * 2);
  _Float16* out_wP = (_Float16*)take((size_t)N_ * H_ * FEAT_ * 2);

  // adjacency: straight f32 -> f16 stream (fits in 192MB L2 afterwards)
  k_cvt<<<(B_ * E_ * N_ * N_) / 2048, 256, 0, stream>>>(adj, adjH,
                                                        B_ * E_ * N_ * N_);
  // weights: convert + pack into WMMA B-fragment panel layout
  {
    int t;
    t = (H_ / 32) * (H_ / 16) * 32 * (L_ * E_);
    k_pack_b<H_, H_, 0><<<(t + 255) / 256, 256, 0, stream>>>(gc_w, gc_wP, L_ * E_);
    t = (H_ / 32) * (3 * H_ / 16) * 32;
    k_pack_b<H_, 3 * H_, 1><<<(t + 255) / 256, 256, 0, stream>>>(w_ih, w_ihP, 1);
    k_pack_b<H_, 3 * H_, 1><<<(t + 255) / 256, 256, 0, stream>>>(w_hh, w_hhP, 1);
    t = (FEAT_ / 32) * (H_ / 16) * 32 * N_;
    k_pack_b<FEAT_, H_, 0><<<(t + 255) / 256, 256, 0, stream>>>(in_w, in_wP, N_);
    t = (H_ / 32) * (FEAT_ / 16) * 32 * N_;
    k_pack_b<H_, FEAT_, 0><<<(t + 255) / 256, 256, 0, stream>>>(out_w, out_wP, N_);
  }

  // ping-pong: embed->out, L0 out->ws, L1 ws->out, L2 out->ws, project ws->out
  k_embed<<<N_ * (B_ / 16), 256, 0, stream>>>(x, in_wP, in_b, out);

  const float* layer_in[L_]  = {out, stateA, out};
  float*       layer_out[L_] = {stateA, out, stateA};
  for (int l = 0; l < L_; ++l) {
    k_layer<<<B_, 256, 0, stream>>>(layer_in[l], adjH,
                                    gc_wP + (size_t)l * E_ * H_ * H_,
                                    gc_b + (size_t)l * E_ * H_,
                                    w_ihP, w_hhP, b_ih, b_hh, layer_out[l]);
  }

  k_out<<<N_ * (B_ / 16), 64, 0, stream>>>(stateA, out_wP, out_b, out);
}
